// MultiInterestExtractor_31997506355613
// MI455X (gfx1250) — compile-verified, matrix operand order checked by reading, not run on hardware
//
#include <hip/hip_runtime.h>
#include <hip/hip_bf16.h>
#include <math.h>

#define B_ 64
#define S_ 200
#define H_ 256
#define A_ 16
#define NEGV (-1e9f)

typedef float    v2f  __attribute__((ext_vector_type(2)));
typedef float    v8f  __attribute__((ext_vector_type(8)));
typedef _Float16 v16h __attribute__((ext_vector_type(16)));

#if defined(__has_builtin)
#if __has_builtin(__builtin_amdgcn_wmma_f32_16x16x4_f32)
#define USE_F32_WMMA 1
#endif
#endif

#ifdef USE_F32_WMMA
// ---------------- native FP32 WMMA path: V_WMMA_F32_16X16X4_F32 ----------------
#define KSTEP 4
#define SPAD 208   // S=200 padded to multiple of KSTEP
typedef v2f afrag;
typedef v2f bfrag;
// A 16x4 f32 layout (ISA 7.12.2): lane m=l%16, VGPR0/1 = K0/K1 (lanes<16) or K2/K3
__device__ __forceinline__ afrag load_a_rm(const float* p, int ld, int kb, int lane) {
  int m = lane & 15;
  int k0 = kb + ((lane < 16) ? 0 : 2);
  afrag a; a.x = p[m * ld + k0]; a.y = p[m * ld + k0 + 1]; return a;
}
// B 4x16, B[k][n] stored row-major at p[k*ld+n]
__device__ __forceinline__ bfrag load_b_rm(const float* p, int ld, int kb, int lane) {
  int n = lane & 15;
  int k0 = kb + ((lane < 16) ? 0 : 2);
  bfrag b; b.x = p[k0 * ld + n]; b.y = p[(k0 + 1) * ld + n]; return b;
}
// B[k][n] stored as p[n*ld+k] (transposed source, e.g. aspect_embs (A,H))
__device__ __forceinline__ bfrag load_b_cm(const float* p, int ld, int kb, int lane) {
  int n = lane & 15;
  int k0 = kb + ((lane < 16) ? 0 : 2);
  bfrag b; b.x = p[n * ld + k0]; b.y = p[n * ld + k0 + 1]; return b;
}
__device__ __forceinline__ v8f wmma_acc(afrag a, bfrag b, v8f c) {
  return __builtin_amdgcn_wmma_f32_16x16x4_f32(false, a, false, b, (short)0, c,
                                               false, false);
}
#else
// ---------------- fallback: confirmed V_WMMA_F32_16X16X32_F16 ----------------
#define KSTEP 32
#define SPAD 224   // S=200 padded to multiple of KSTEP
typedef v16h afrag;
typedef v16h bfrag;
// 16-bit 16x32 operand layout (ISA 7.12.2): half j -> K = (j%8) + (j/8)*16 (+8 hi lanes)
__device__ __forceinline__ int kmap16(int j, int lane) {
  return (j & 7) + ((j >> 3) << 4) + ((lane < 16) ? 0 : 8);
}
__device__ __forceinline__ afrag load_a_rm(const float* p, int ld, int kb, int lane) {
  int m = lane & 15;
  afrag a;
#pragma unroll
  for (int j = 0; j < 16; ++j) a[j] = (_Float16)p[m * ld + kb + kmap16(j, lane)];
  return a;
}
__device__ __forceinline__ bfrag load_b_rm(const float* p, int ld, int kb, int lane) {
  int n = lane & 15;
  bfrag b;
#pragma unroll
  for (int j = 0; j < 16; ++j) b[j] = (_Float16)p[(kb + kmap16(j, lane)) * ld + n];
  return b;
}
__device__ __forceinline__ bfrag load_b_cm(const float* p, int ld, int kb, int lane) {
  int n = lane & 15;
  bfrag b;
#pragma unroll
  for (int j = 0; j < 16; ++j) b[j] = (_Float16)p[n * ld + kb + kmap16(j, lane)];
  return b;
}
__device__ __forceinline__ v8f wmma_acc(afrag a, bfrag b, v8f c) {
  return __builtin_amdgcn_wmma_f32_16x16x32_f16(false, a, false, b, (short)0, c,
                                                false, false);
}
#endif

__device__ __forceinline__ v8f vzero() {
  v8f z;
#pragma unroll
  for (int i = 0; i < 8; ++i) z[i] = 0.f;
  return z;
}
// C/D 16x16 f32 layout: VGPR r: lanes 0-15 -> M=r, lanes 16-31 -> M=8+r; N=lane%16
__device__ __forceinline__ int c_row(int r, int lane) { return r + ((lane < 16) ? 0 : 8); }
__device__ __forceinline__ int c_col(int lane) { return lane & 15; }

// ---------------------------------------------------------------------------
// K0: zero counts and the 32 zero-pad rows appended to moe (K-padding safety)
__global__ void init_kernel(float* __restrict__ moe_pad, float* __restrict__ counts) {
  int i = blockIdx.x * 256 + threadIdx.x;
  if (i < 32 * H_) moe_pad[i] = 0.f;
  if (i < B_ * A_) counts[i] = 0.f;
}

// K1: moe = LayerNorm(tanh(item_emb @ W_lin + b_lin) + item_emb)
__global__ void moe_kernel(const float* __restrict__ item_emb,
                           const float* __restrict__ W_lin,
                           const float* __restrict__ b_lin,
                           const float* __restrict__ ln_g,
                           const float* __restrict__ ln_b,
                           float* __restrict__ moe) {
  __shared__ float Atile[16 * H_];
  __shared__ float red1[16], red2[16];
  const int tid = threadIdx.x, lane = tid & 31, wave = tid >> 5;
  const int row0 = blockIdx.x * 16;
  for (int i = tid; i < 16 * H_; i += 256) Atile[i] = item_emb[(size_t)row0 * H_ + i];
  if (tid < 16) { red1[tid] = 0.f; red2[tid] = 0.f; }
  __syncthreads();

  v8f acc[2]; acc[0] = vzero(); acc[1] = vzero();
  for (int kb = 0; kb < H_; kb += KSTEP) {
    afrag a = load_a_rm(Atile, H_, kb, lane);
#pragma unroll
    for (int t = 0; t < 2; ++t) {
      int n0 = (wave * 2 + t) * 16;
      bfrag bf = load_b_rm(W_lin + n0, H_, kb, lane);
      acc[t] = wmma_acc(a, bf, acc[t]);
    }
  }
  __syncthreads();  // all reads of Atile done; safe to overwrite
#pragma unroll
  for (int t = 0; t < 2; ++t) {
    int n0 = (wave * 2 + t) * 16;
#pragma unroll
    for (int r = 0; r < 8; ++r) {
      int row = c_row(r, lane), col = n0 + c_col(lane);
      float tv = acc[t][r] + b_lin[col];
      Atile[row * H_ + col] = tanhf(tv) + Atile[row * H_ + col];
    }
  }
  __syncthreads();
  {
    int row = tid >> 4, j = tid & 15;
    float s1 = 0.f, s2 = 0.f;
    for (int c = j; c < H_; c += 16) { float v = Atile[row * H_ + c]; s1 += v; s2 += v * v; }
    atomicAdd(&red1[row], s1); atomicAdd(&red2[row], s2);
  }
  __syncthreads();
  {
    int row = tid >> 4, j = tid & 15;
    float mu = red1[row] * (1.f / H_);
    float var = red2[row] * (1.f / H_) - mu * mu;
    float rstd = rsqrtf(var + 1e-12f);
    for (int c = j; c < H_; c += 16) {
      float v = Atile[row * H_ + c];
      moe[(size_t)(row0 + row) * H_ + c] = (v - mu) * rstd * ln_g[c] + ln_b[c];
    }
  }
}

// K2: gates = item_emb @ aspect_embs^T; fused argmax/softmax/bij-init/counts
__global__ void gates_kernel(const float* __restrict__ item_emb,
                             const int* __restrict__ item_seq,
                             const float* __restrict__ aspect_embs,
                             float* __restrict__ gates_sm,
                             float* __restrict__ idx_out_f,
                             int* __restrict__ idx_ws,
                             float* __restrict__ bij,
                             float* __restrict__ counts) {
  __shared__ float gt[64][A_];
  const int tid = threadIdx.x, lane = tid & 31, wave = tid >> 5;
  const int row0 = blockIdx.x * 64 + wave * 16;

  v8f acc = vzero();
  for (int kb = 0; kb < H_; kb += KSTEP) {
    afrag a = load_a_rm(item_emb + (size_t)row0 * H_, H_, kb, lane);
    bfrag bf = load_b_cm(aspect_embs, H_, kb, lane);  // B[k][n] = aspect_embs[n][k]
    acc = wmma_acc(a, bf, acc);
  }
#pragma unroll
  for (int r = 0; r < 8; ++r) gt[wave * 16 + c_row(r, lane)][c_col(lane)] = acc[r];
  __syncthreads();

  if (tid < 64) {
    int grow = blockIdx.x * 64 + tid;  // flattened (b*S+s)
    float mx = -INFINITY; int am = 0;
    for (int a = 0; a < A_; ++a) { float v = gt[tid][a]; if (v > mx) { mx = v; am = a; } }
    float ex[A_], se = 0.f;
    for (int a = 0; a < A_; ++a) { ex[a] = __expf(gt[tid][a] - mx); se += ex[a]; }
    float inv = 1.f / se;
    for (int a = 0; a < A_; ++a) {
      gates_sm[(size_t)grow * A_ + a] = ex[a] * inv;
      bij[(size_t)grow * A_ + a] = gt[tid][a];
    }
    idx_ws[grow] = am;
    idx_out_f[grow] = (float)am;
    int b = grow / S_;
    float src = (item_seq[grow] == 0) ? 0.f : 1.f;
    atomicAdd(&counts[b * A_ + am], src);
  }
}

// K3: w = LeakyReLU((item+pos+td)@Wa1+ba1)@Wa2 + ba2, pad -> NEG (no h materialized)
__global__ void tma_w_kernel(const float* __restrict__ item_emb,
                             const int* __restrict__ item_seq,
                             const int* __restrict__ time_seq,
                             const float* __restrict__ pos_emb,
                             const float* __restrict__ tdt,
                             const float* __restrict__ tdn,
                             const float* __restrict__ Wa1,
                             const float* __restrict__ ba1,
                             const float* __restrict__ Wa2,
                             const float* __restrict__ ba2,
                             float* __restrict__ w_out) {
  __shared__ float Atile[16 * H_];
  __shared__ float wred[16];
  const int tid = threadIdx.x, lane = tid & 31, wave = tid >> 5;
  const int row0 = blockIdx.x * 16;
  for (int i = tid; i < 16 * H_; i += 256) {
    int row = i >> 8, col = i & 255;
    int g = row0 + row, s = g % S_;
    int t0 = time_seq[g * 2 + 0], t1 = time_seq[g * 2 + 1];
    Atile[i] = item_emb[(size_t)g * H_ + col] + pos_emb[s * H_ + col] +
               tdt[t0 * H_ + col] + tdn[t1 * H_ + col];
  }
  if (tid < 16) wred[tid] = 0.f;
  __syncthreads();

  v8f acc[2]; acc[0] = vzero(); acc[1] = vzero();
  for (int kb = 0; kb < H_; kb += KSTEP) {
    afrag a = load_a_rm(Atile, H_, kb, lane);
#pragma unroll
    for (int t = 0; t < 2; ++t) {
      int n0 = (wave * 2 + t) * 16;
      bfrag bf = load_b_rm(Wa1 + n0, H_, kb, lane);
      acc[t] = wmma_acc(a, bf, acc[t]);
    }
  }
#pragma unroll
  for (int t = 0; t < 2; ++t) {
    int n0 = (wave * 2 + t) * 16;
#pragma unroll
    for (int r = 0; r < 8; ++r) {
      int row = c_row(r, lane), col = n0 + c_col(lane);
      float h = acc[t][r] + ba1[col];
      h = (h >= 0.f) ? h : 0.01f * h;
      atomicAdd(&wred[row], h * Wa2[col]);
    }
  }
  __syncthreads();
  if (tid < 16) {
    int g = row0 + tid;
    float wv = wred[tid] + ba2[0];
    if (item_seq[g] == 0) wv = NEGV;
    w_out[g] = wv;
  }
}

// K4: tma = softmax over S per batch (in-place on w)
__global__ void tma_softmax_kernel(float* __restrict__ w) {
  __shared__ float red[256];
  const int b = blockIdx.x, tid = threadIdx.x;
  float v = (tid < S_) ? w[b * S_ + tid] : -INFINITY;
  red[tid] = v; __syncthreads();
  for (int o = 128; o > 0; o >>= 1) { if (tid < o) red[tid] = fmaxf(red[tid], red[tid + o]); __syncthreads(); }
  float mx = red[0]; __syncthreads();
  float e = (tid < S_) ? __expf(v - mx) : 0.f;
  red[tid] = e; __syncthreads();
  for (int o = 128; o > 0; o >>= 1) { if (tid < o) red[tid] += red[tid + o]; __syncthreads(); }
  float inv = 1.f / red[0];
  if (tid < S_) w[b * S_ + tid] = e * inv;
}

// K5: aspect_mask = (counts == 0)
__global__ void mask_kernel(const float* __restrict__ counts,
                            float* __restrict__ mask_out, int* __restrict__ mask_ws) {
  int i = blockIdx.x * blockDim.x + threadIdx.x;
  if (i < B_ * A_) {
    int m = (counts[i] == 0.f) ? 1 : 0;
    mask_out[i] = (float)m;
    mask_ws[i] = m;
  }
}

// K6: one full routing layer per batch per block (cij softmax, cap GEMM, squash, bij GEMM)
__global__ void routing_kernel(const float* __restrict__ moe,
                               const float* __restrict__ tma,
                               const int* __restrict__ item_seq,
                               const int* __restrict__ mask_ws,
                               float* __restrict__ bij,
                               float* __restrict__ cap_out,
                               int update_bij) {
  __shared__ float Acap[A_ * SPAD];  // A[a][s] = cij[s][a]*tma[s], K-padded with zeros
  __shared__ float capS[A_ * H_];
  __shared__ float n2red[A_];
  const int b = blockIdx.x, tid = threadIdx.x, lane = tid & 31, wave = tid >> 5;

  for (int i = tid; i < A_ * SPAD; i += 256) Acap[i] = 0.f;
  if (tid < A_) n2red[tid] = 0.f;
  __syncthreads();

  for (int s = tid; s < S_; s += 256) {
    const float* bp = bij + ((size_t)b * S_ + s) * A_;
    float row[A_]; float mx = -INFINITY;
    for (int a = 0; a < A_; ++a) {
      float v = mask_ws[b * A_ + a] ? NEGV : bp[a];
      row[a] = v; mx = fmaxf(mx, v);
    }
    float se = 0.f;
    for (int a = 0; a < A_; ++a) { row[a] = __expf(row[a] - mx); se += row[a]; }
    float scale = (item_seq[b * S_ + s] == 0) ? 0.f : (1.f / se);
    float tv = tma[b * S_ + s] * scale;
    for (int a = 0; a < A_; ++a) Acap[a * SPAD + s] = row[a] * tv;
  }
  __syncthreads();

  // cap[a][h] = sum_s Acap[a][s] * moe[b][s][h]
  const float* moe_b = moe + (size_t)b * S_ * H_;
  v8f acc[2]; acc[0] = vzero(); acc[1] = vzero();
  for (int kb = 0; kb < SPAD; kb += KSTEP) {
    afrag a = load_a_rm(Acap, SPAD, kb, lane);
#pragma unroll
    for (int t = 0; t < 2; ++t) {
      int n0 = (wave * 2 + t) * 16;
      bfrag bf = load_b_rm(moe_b + n0, H_, kb, lane);  // K rows >= S hit zero-pad rows
      acc[t] = wmma_acc(a, bf, acc[t]);
    }
  }
#pragma unroll
  for (int t = 0; t < 2; ++t) {
    int n0 = (wave * 2 + t) * 16;
#pragma unroll
    for (int r = 0; r < 8; ++r) capS[c_row(r, lane) * H_ + n0 + c_col(lane)] = acc[t][r];
  }
  __syncthreads();

  // squash
  {
    int row = tid >> 4, j = tid & 15;
    float s2 = 0.f;
    for (int c = j; c < H_; c += 16) { float v = capS[row * H_ + c]; s2 += v * v; }
    atomicAdd(&n2red[row], s2);
  }
  __syncthreads();
  {
    int row = tid >> 4, j = tid & 15;
    float n2 = n2red[row];
    float sc = n2 / ((1.f + n2) * sqrtf(n2 + 1e-9f));
    for (int c = j; c < H_; c += 16) {
      float v = capS[row * H_ + c] * sc;
      capS[row * H_ + c] = v;
      cap_out[((size_t)b * A_ + row) * H_ + c] = v;
    }
  }
  __syncthreads();

  if (update_bij) {
    // bij[s][a] += sum_h moe[b][s][h] * cap[a][h]; 13 row tiles of 16 over S(+pad)
    for (int tt = wave; tt < 13; tt += 8) {
      v8f acc2 = vzero();
      const float* arow = moe + ((size_t)b * S_ + tt * 16) * H_;
      for (int kb = 0; kb < H_; kb += KSTEP) {
        afrag a = load_a_rm(arow, H_, kb, lane);
        bfrag bf = load_b_cm(capS, H_, kb, lane);  // B[k=h][n=a] = capS[a][h]
        acc2 = wmma_acc(a, bf, acc2);
      }
#pragma unroll
      for (int r = 0; r < 8; ++r) {
        int s = tt * 16 + c_row(r, lane);
        if (s < S_) bij[((size_t)b * S_ + s) * A_ + c_col(lane)] += acc2[r];
      }
    }
  }
}

// K7: activated[b,s,:] = cap[b, idx[b,s], :]
__global__ void activated_kernel(const float* __restrict__ cap,
                                 const int* __restrict__ idx,
                                 float* __restrict__ out) {
  size_t i = (size_t)blockIdx.x * 256 + threadIdx.x;  // over B*S*H
  int row = (int)(i >> 8);
  int col = (int)(i & 255);
  int b = row / S_;
  int a = idx[row];
  out[i] = cap[((size_t)b * A_ + a) * H_ + col];
}

// K8: aspects_item_idx
__global__ void lastpos_kernel(const int* __restrict__ idx,
                               const int* __restrict__ item_seq,
                               float* __restrict__ out) {
  int i = blockIdx.x * blockDim.x + threadIdx.x;
  if (i >= B_ * A_) return;
  int b = i / A_, a = i % A_;
  float mp = -1.f;
  for (int s = 0; s < S_; ++s)
    if (item_seq[b * S_ + s] != 0 && idx[b * S_ + s] == a) mp = (float)(s + 1);
  out[i] = (mp > -1.f) ? mp - 1.f : mp;
}

// ---------------------------------------------------------------------------
extern "C" void kernel_launch(void* const* d_in, const int* in_sizes, int n_in,
                              void* d_out, int out_size, void* d_ws, size_t ws_size,
                              hipStream_t stream) {
  const float* item_emb    = (const float*)d_in[0];
  const int*   item_seq    = (const int*)d_in[1];
  const int*   time_seq    = (const int*)d_in[2];
  const float* W_lin       = (const float*)d_in[3];
  const float* b_lin       = (const float*)d_in[4];
  const float* aspect_embs = (const float*)d_in[5];
  const float* pos_emb     = (const float*)d_in[6];
  const float* tdt         = (const float*)d_in[7];
  const float* tdn         = (const float*)d_in[8];
  const float* Wa1         = (const float*)d_in[9];
  const float* ba1         = (const float*)d_in[10];
  const float* Wa2         = (const float*)d_in[11];
  const float* ba2         = (const float*)d_in[12];
  const float* ln_g        = (const float*)d_in[13];
  const float* ln_b        = (const float*)d_in[14];

  float* out       = (float*)d_out;
  float* cap_out   = out;                                   // B*A*H
  float* gates_sm  = cap_out + (size_t)B_ * A_ * H_;        // B*S*A
  float* mask_out  = gates_sm + (size_t)B_ * S_ * A_;       // B*A
  float* activated = mask_out + B_ * A_;                    // B*S*H
  float* lastpos   = activated + (size_t)B_ * S_ * H_;      // B*A
  float* idx_out_f = lastpos + B_ * A_;                     // B*S

  float* moe    = (float*)d_ws;                             // (B*S+32)*H
  float* wbuf   = moe + (size_t)(B_ * S_ + 32) * H_;        // B*S (w -> tma)
  float* bij    = wbuf + B_ * S_;                           // B*S*A
  float* counts = bij + (size_t)B_ * S_ * A_;               // B*A
  int*   mask_ws = (int*)(counts + B_ * A_);                // B*A
  int*   idx_ws  = mask_ws + B_ * A_;                       // B*S

  init_kernel<<<32, 256, 0, stream>>>(moe + (size_t)B_ * S_ * H_, counts);
  moe_kernel<<<B_ * S_ / 16, 256, 0, stream>>>(item_emb, W_lin, b_lin, ln_g, ln_b, moe);
  gates_kernel<<<B_ * S_ / 64, 128, 0, stream>>>(item_emb, item_seq, aspect_embs,
                                                 gates_sm, idx_out_f, idx_ws, bij, counts);
  tma_w_kernel<<<B_ * S_ / 16, 256, 0, stream>>>(item_emb, item_seq, time_seq, pos_emb,
                                                 tdt, tdn, Wa1, ba1, Wa2, ba2, wbuf);
  tma_softmax_kernel<<<B_, 256, 0, stream>>>(wbuf);
  mask_kernel<<<4, 256, 0, stream>>>(counts, mask_out, mask_ws);
  for (int it = 0; it < 3; ++it)
    routing_kernel<<<B_, 256, 0, stream>>>(moe, wbuf, item_seq, mask_ws, bij, cap_out,
                                           (it < 2) ? 1 : 0);
  activated_kernel<<<B_ * S_, 256, 0, stream>>>(cap_out, idx_ws, activated);
  lastpos_kernel<<<4, 256, 0, stream>>>(idx_ws, item_seq, lastpos);
}
